// RNN_18502719111330
// MI455X (gfx1250) — matrix operand order repaired and checked by reading, not compiled, not run on hardware
//
#include <hip/hip_runtime.h>
#include <math.h>

typedef __attribute__((ext_vector_type(16))) _Float16 v16h;
typedef __attribute__((ext_vector_type(8)))  float    v8f;

#define SEQ_L  256
#define BATCH  4096
#define E_DIM  80
#define H_DIM  50

// Hardware tanh (v_tanh_f32 is a TRANS op on gfx1250); fast fallback otherwise.
__device__ __forceinline__ float fast_tanh(float x) {
#if __has_builtin(__builtin_amdgcn_tanhf)
  return __builtin_amdgcn_tanhf(x);
#else
  const float ax = __builtin_fabsf(x);
  const float e  = __builtin_amdgcn_exp2f(ax * -2.885390081777927f);
  const float r  = (1.0f - e) * __builtin_amdgcn_rcpf(1.0f + e);
  return __builtin_copysignf(r, x);
#endif
}

// One wave (32 threads) per block; each wave owns 16 batch rows for all timesteps.
__global__ __launch_bounds__(32) void rnn_elman_wmma(
    const float* __restrict__ x,    // [L, B, E]
    const float* __restrict__ Wi,   // [E, H]
    const float* __restrict__ Wh,   // [H, H]
    const float* __restrict__ bh,   // [H]
    float* __restrict__ out)        // [L, B, H]
{
  __shared__ _Float16 hlds[16 * 64];   // 2 KB: h-tile transpose staging (C-layout -> A-layout)

  const int lane = threadIdx.x & 31;
  const int m    = lane & 15;          // row / column index within 16
  const int hi   = lane >> 4;          // lane half select
  const int rowbase = blockIdx.x * 16; // batch rows [rowbase, rowbase+16)

  // ---- Preload B-fragments (16-bit B 32x16 layout: per lane col n = nt*16 + m,
  //      element j holds k = c*32 + ((j&8)<<1) + hi*8 + (j&7)) ----
  v16h WiB[3][4];
  v16h WhB[2][4];
#pragma unroll
  for (int c = 0; c < 3; ++c) {
#pragma unroll
    for (int nt = 0; nt < 4; ++nt) {
      v16h f;
      const int n = nt * 16 + m;
#pragma unroll
      for (int j = 0; j < 16; ++j) {
        const int k = c * 32 + ((j & 8) << 1) + hi * 8 + (j & 7);
        f[j] = (k < E_DIM && n < H_DIM) ? (_Float16)Wi[k * H_DIM + n] : (_Float16)0.0f;
      }
      WiB[c][nt] = f;
    }
  }
  // Wh with the bias folded into padded row k = H_DIM (h's constant-1 column):
#pragma unroll
  for (int c = 0; c < 2; ++c) {
#pragma unroll
    for (int nt = 0; nt < 4; ++nt) {
      v16h f;
      const int n = nt * 16 + m;
#pragma unroll
      for (int j = 0; j < 16; ++j) {
        const int k = c * 32 + ((j & 8) << 1) + hi * 8 + (j & 7);
        float w = 0.0f;
        if (n < H_DIM) {
          if (k < H_DIM)       w = Wh[k * H_DIM + n];
          else if (k == H_DIM) w = bh[n];            // bias row
        }
        f[j] = (_Float16)w;
      }
      WhB[c][nt] = f;
    }
  }

  // h_0 = 0, except the constant-1 column at k = H_DIM (bias trick).
  v16h hA[2];
#pragma unroll
  for (int c = 0; c < 2; ++c) {
#pragma unroll
    for (int j = 0; j < 16; ++j) {
      const int k = c * 32 + ((j & 8) << 1) + hi * 8 + (j & 7);
      hA[c][j] = (k == H_DIM) ? (_Float16)1.0f : (_Float16)0.0f;
    }
  }

  // Per-lane base pointers, advanced by one timestep each iteration.
  // Output: element offset for (v, nt) from out_lane is the constant v*H_DIM + nt*16.
  const float* xrow = x + (size_t)(rowbase + m) * E_DIM;
  float* out_lane   = out + (size_t)rowbase * H_DIM + hi * 8 * H_DIM + m;

  for (int t = 0; t < SEQ_L; ++t) {
    // C starts at zero: bias arrives through the Wh WMMA via h's constant column.
    v8f C[4];
#pragma unroll
    for (int nt = 0; nt < 4; ++nt) {
      v8f z = {};
      C[nt] = z;
    }

    // ---- x_t contribution: A = f16(x[t, rowbase+m, :]) per 16-bit 16x32 A layout ----
#pragma unroll
    for (int c = 0; c < 3; ++c) {
      v16h a;
#pragma unroll
      for (int j = 0; j < 16; ++j) {
        const int k = c * 32 + ((j & 8) << 1) + hi * 8 + (j & 7);
        a[j] = (k < E_DIM) ? (_Float16)xrow[k] : (_Float16)0.0f;
      }
#pragma unroll
      for (int nt = 0; nt < 4; ++nt)
        C[nt] = __builtin_amdgcn_wmma_f32_16x16x32_f16(
            false, a, false, WiB[c][nt], (short)0, C[nt], false, false);
    }

    // ---- h_{t-1} @ Wh contribution (includes bias via constant-1 column) ----
#pragma unroll
    for (int c = 0; c < 2; ++c)
#pragma unroll
      for (int nt = 0; nt < 4; ++nt)
        C[nt] = __builtin_amdgcn_wmma_f32_16x16x32_f16(
            false, hA[c], false, WhB[c][nt], (short)0, C[nt], false, false);

    // ---- tanh, write hs (f32, constant-offset stores), stage new h to LDS ----
#pragma unroll
    for (int nt = 0; nt < 4; ++nt) {
      const int n = nt * 16 + m;     // C layout: N = lane&15
#pragma unroll
      for (int v = 0; v < 8; ++v) {
        const int mm = v + hi * 8;   // C layout: M = vgpr (+8 for lanes 16-31)
        const float hv = fast_tanh(C[nt][v]);
        if (n < H_DIM) out_lane[v * H_DIM + nt * 16] = hv;  // folds into IOFFSET
        // padded cols: n==H_DIM must stay the constant 1.0 (bias column), rest 0
        hlds[mm * 64 + n] = (n == H_DIM) ? (_Float16)1.0f : (_Float16)hv;
      }
    }

    // DS ops are in-order within a wave; wait before the transposed reload.
    asm volatile("s_wait_dscnt 0" ::: "memory");

    // ---- reload h as A-fragments (16-bit 16x32 A layout) ----
#pragma unroll
    for (int c = 0; c < 2; ++c) {
      v16h a;
#pragma unroll
      for (int j = 0; j < 16; ++j) {
        const int k = c * 32 + ((j & 8) << 1) + hi * 8 + (j & 7);
        a[j] = hlds[m * 64 + k];
      }
      hA[c] = a;
    }

    // Advance per-lane pointers one timestep.
    xrow     += (size_t)BATCH * E_DIM;
    out_lane += (size_t)BATCH * H_DIM;
  }
}

extern "C" void kernel_launch(void* const* d_in, const int* in_sizes, int n_in,
                              void* d_out, int out_size, void* d_ws, size_t ws_size,
                              hipStream_t stream) {
  const float* x  = (const float*)d_in[0];
  const float* Wi = (const float*)d_in[1];
  const float* Wh = (const float*)d_in[2];
  const float* bh = (const float*)d_in[3];
  float* out = (float*)d_out;
  (void)in_sizes; (void)n_in; (void)out_size; (void)d_ws; (void)ws_size;

  dim3 grid(BATCH / 16);   // 256 blocks, one wave32 each
  dim3 block(32);
  hipLaunchKernelGGL(rnn_elman_wmma, grid, block, 0, stream, x, Wi, Wh, bh, out);
}
